// CGRM_27315992002867
// MI455X (gfx1250) — matrix operand order repaired and checked by reading, not compiled, not run on hardware
//
#include <hip/hip_runtime.h>
#include <math.h>

// ---------------------------------------------------------------------------
// Problem constants (match reference): B=8, T=2048, D=768, WINDOW=8, STRIDE=4
//   events M = len(range(0, T-WINDOW+1, STRIDE)) = 511
// All f32 in/out; GEMMs run through v_wmma_f32_16x16x32_f16 (f16 in, f32 acc).
// ---------------------------------------------------------------------------
#define B_   8
#define T_   2048
#define DIM  768
#define M_   511

typedef __attribute__((ext_vector_type(16))) _Float16 v16h;
typedef __attribute__((ext_vector_type(8)))  float    v8f;

// membership helpers: event i starts at 4i, covers frames [4i, 4i+8)
__device__ __forceinline__ int row_lo(int t) { return (t <= 6) ? 0 : ((t - 4) >> 2); }
__device__ __forceinline__ int row_hi(int t) { int h = t >> 2; return h > (M_ - 1) ? (M_ - 1) : h; }
__device__ __forceinline__ float gelu_exact(float x) {
  return 0.5f * x * (1.0f + erff(x * 0.70710678118654752f));
}

// ---------------------------------------------------------------------------
// Generic fused WMMA GEMM:  out = act( scale * (A' x B') + bias ) [* mask]
//   A' = rowscale[m] * A[m,k]        (rowscale optional -> norm_adj / cosine)
//   B' = colscale[n] * (transB ? Bm[n,k] : Bm[k,n])
// Block tile 128x128, K-step 32, 8 waves (4 row-strips x 2 col-strips);
// each wave owns a 32x64 tile = 2x4 of 16x16x32 WMMA (8 WMMA / iter / wave).
// Register-pipelined staging: global loads of tile k+1 overlap WMMA on tile k.
// Block-uniform fast/slow path selection keeps the hot loop branchless.
// ---------------------------------------------------------------------------
#define BLK_M 128
#define BLK_N 128
#define BLK_K 32
#define LDS_K (BLK_K + 8)   // half-stride pad vs bank conflicts

__global__ __launch_bounds__(256)
void wmma_gemm_kernel(const float* __restrict__ A, long long sA, int lda,
                      const float* __restrict__ Bm, long long sB, int ldb, int transB,
                      const float* __restrict__ bias,
                      const float* __restrict__ rowscale, long long sRS,
                      const float* __restrict__ colscale, long long sCS,
                      float* __restrict__ out, long long sOut, int ldo,
                      int Mrows, int K, int Ncols,
                      float scale, int act, int strictUpper)
{
  __shared__ _Float16 As[BLK_M][LDS_K];
  __shared__ _Float16 Bs[BLK_N][LDS_K];   // stored [n][k] for both NN / NT

  const int tid = threadIdx.x;
  const int bz = blockIdx.z;
  const int rowBase = blockIdx.y * BLK_M;
  const int colBase = blockIdx.x * BLK_N;

  const float* Ab  = A  + (long long)bz * sA;
  const float* Bb  = Bm + (long long)bz * sB;
  const float* rsb = rowscale ? rowscale + (long long)bz * sRS : nullptr;
  const float* csb = colscale ? colscale + (long long)bz * sCS : nullptr;
  float*       Ob  = out + (long long)bz * sOut;

  const int wave = tid >> 5;
  const int lane = tid & 31;
  const int wrow = wave & 3;          // 4 wave-rows  (32 rows each)
  const int wcol = wave >> 2;         // 2 wave-cols  (64 cols each)
  const int lr = lane & 15;
  const int lh = lane >> 4;

  const bool alA = ((lda & 3) == 0);
  const bool alB = ((ldb & 3) == 0);
  const bool fullM = (rowBase + BLK_M <= Mrows);
  const bool fullN = (colBase + BLK_N <= Ncols);

  // --- staging helpers: global -> registers (scaled), registers -> LDS(f16)
  // A tile: 128 rows x 32 k = 1024 float4, 4 per thread; i = tid + j*256
  //   r = i>>3, k4 = (i&7)*4
  auto loadA = [&](int kb, float4* reg) {
    const bool fast = alA && fullM && (kb + BLK_K <= K);   // block-uniform
    if (fast) {
#pragma unroll
      for (int j = 0; j < 4; ++j) {
        int i = tid + j * 256;
        int r = i >> 3, k4 = (i & 7) << 2;
        int gr = rowBase + r;
        float4 v = *(const float4*)(Ab + (long long)gr * lda + kb + k4);
        if (rsb) { float s = rsb[gr]; v.x *= s; v.y *= s; v.z *= s; v.w *= s; }
        reg[j] = v;
      }
    } else {
#pragma unroll
      for (int j = 0; j < 4; ++j) {
        int i = tid + j * 256;
        int r = i >> 3, k4 = (i & 7) << 2;
        int gr = rowBase + r;
        float4 v = {0.f, 0.f, 0.f, 0.f};
        if (gr < Mrows) {
          const float* p = Ab + (long long)gr * lda + kb + k4;
          int rem = K - kb - k4;
          if (rem > 0) v.x = p[0];
          if (rem > 1) v.y = p[1];
          if (rem > 2) v.z = p[2];
          if (rem > 3) v.w = p[3];
          if (rsb) { float s = rsb[gr]; v.x *= s; v.y *= s; v.z *= s; v.w *= s; }
        }
        reg[j] = v;
      }
    }
  };
  auto storeA = [&](const float4* reg) {
#pragma unroll
    for (int j = 0; j < 4; ++j) {
      int i = tid + j * 256;
      int r = i >> 3, k4 = (i & 7) << 2;
      As[r][k4 + 0] = (_Float16)reg[j].x; As[r][k4 + 1] = (_Float16)reg[j].y;
      As[r][k4 + 2] = (_Float16)reg[j].z; As[r][k4 + 3] = (_Float16)reg[j].w;
    }
  };
  // B tile -> Bs[n][k] (128 x 32)
  //   NN: i -> k = i>>5, n4 = (i&31)*4  (read along n, contiguous)
  //   NT: i -> n = i>>3, k4 = (i&7)*4   (read along k, contiguous)
  auto loadB = [&](int kb, float4* reg) {
    if (!transB) {
      const bool fast = alB && fullN && (kb + BLK_K <= K);
      if (fast) {
#pragma unroll
        for (int j = 0; j < 4; ++j) {
          int i = tid + j * 256;
          int k = i >> 5, n4 = (i & 31) << 2;
          float4 v = *(const float4*)(Bb + (long long)(kb + k) * ldb + colBase + n4);
          if (csb) {
            v.x *= csb[colBase + n4 + 0]; v.y *= csb[colBase + n4 + 1];
            v.z *= csb[colBase + n4 + 2]; v.w *= csb[colBase + n4 + 3];
          }
          reg[j] = v;
        }
      } else {
#pragma unroll
        for (int j = 0; j < 4; ++j) {
          int i = tid + j * 256;
          int k = i >> 5, n4 = (i & 31) << 2;
          int gk = kb + k;
          float4 v = {0.f, 0.f, 0.f, 0.f};
          if (gk < K) {
            const float* p = Bb + (long long)gk * ldb + colBase + n4;
            if (colBase + n4 + 0 < Ncols) v.x = p[0];
            if (colBase + n4 + 1 < Ncols) v.y = p[1];
            if (colBase + n4 + 2 < Ncols) v.z = p[2];
            if (colBase + n4 + 3 < Ncols) v.w = p[3];
            if (csb) {
              if (colBase + n4 + 0 < Ncols) v.x *= csb[colBase + n4 + 0];
              if (colBase + n4 + 1 < Ncols) v.y *= csb[colBase + n4 + 1];
              if (colBase + n4 + 2 < Ncols) v.z *= csb[colBase + n4 + 2];
              if (colBase + n4 + 3 < Ncols) v.w *= csb[colBase + n4 + 3];
            }
          }
          reg[j] = v;
        }
      }
    } else {
      const bool fast = alB && fullN && (kb + BLK_K <= K);
      if (fast) {
#pragma unroll
        for (int j = 0; j < 4; ++j) {
          int i = tid + j * 256;
          int n = i >> 3, k4 = (i & 7) << 2;
          int gn = colBase + n;
          float4 v = *(const float4*)(Bb + (long long)gn * ldb + kb + k4);
          if (csb) { float s = csb[gn]; v.x *= s; v.y *= s; v.z *= s; v.w *= s; }
          reg[j] = v;
        }
      } else {
#pragma unroll
        for (int j = 0; j < 4; ++j) {
          int i = tid + j * 256;
          int n = i >> 3, k4 = (i & 7) << 2;
          int gn = colBase + n;
          float4 v = {0.f, 0.f, 0.f, 0.f};
          if (gn < Ncols) {
            const float* p = Bb + (long long)gn * ldb + kb + k4;
            int rem = K - kb - k4;
            if (rem > 0) v.x = p[0];
            if (rem > 1) v.y = p[1];
            if (rem > 2) v.z = p[2];
            if (rem > 3) v.w = p[3];
            if (csb) { float s = csb[gn]; v.x *= s; v.y *= s; v.z *= s; v.w *= s; }
          }
          reg[j] = v;
        }
      }
    }
  };
  auto storeB = [&](const float4* reg) {
    if (!transB) {
#pragma unroll
      for (int j = 0; j < 4; ++j) {
        int i = tid + j * 256;
        int k = i >> 5, n4 = (i & 31) << 2;
        Bs[n4 + 0][k] = (_Float16)reg[j].x; Bs[n4 + 1][k] = (_Float16)reg[j].y;
        Bs[n4 + 2][k] = (_Float16)reg[j].z; Bs[n4 + 3][k] = (_Float16)reg[j].w;
      }
    } else {
#pragma unroll
      for (int j = 0; j < 4; ++j) {
        int i = tid + j * 256;
        int n = i >> 3, k4 = (i & 7) << 2;
        Bs[n][k4 + 0] = (_Float16)reg[j].x; Bs[n][k4 + 1] = (_Float16)reg[j].y;
        Bs[n][k4 + 2] = (_Float16)reg[j].z; Bs[n][k4 + 3] = (_Float16)reg[j].w;
      }
    }
  };

  v8f c[2][4] = {};
  float4 aR[4], bR[4];

  loadA(0, aR);
  loadB(0, bR);

  for (int kb = 0; kb < K; kb += BLK_K) {
    storeA(aR);
    storeB(bR);
    __syncthreads();

    // prefetch next tile into registers while this tile computes
    int kn = kb + BLK_K;
    if (kn < K) { loadA(kn, aR); loadB(kn, bR); }

    // ---- fragments per ISA 7.12.2 layouts, then 8 WMMAs ----
    v16h amat[2], bmat[4];
#pragma unroll
    for (int mi = 0; mi < 2; ++mi) {
      int arow = wrow * 32 + mi * 16 + lr;
#pragma unroll
      for (int j = 0; j < 8; ++j) {
        amat[mi][j]     = As[arow][lh * 8 + j];        // K 0..7  / 8..15
        amat[mi][8 + j] = As[arow][16 + lh * 8 + j];   // K 16..23 / 24..31
      }
    }
#pragma unroll
    for (int ni = 0; ni < 4; ++ni) {
      int bn = wcol * 64 + ni * 16 + lr;
#pragma unroll
      for (int j = 0; j < 16; ++j)
        bmat[ni][j] = Bs[bn][(j < 8) ? (lh * 8 + j) : (16 + lh * 8 + (j - 8))];  // K 0..7,16..23 / 8..15,24..31
    }
#pragma unroll
    for (int mi = 0; mi < 2; ++mi)
#pragma unroll
      for (int ni = 0; ni < 4; ++ni)
        c[mi][ni] = __builtin_amdgcn_wmma_f32_16x16x32_f16(
            false, amat[mi], false, bmat[ni], (short)0, c[mi][ni], false, false);

    __syncthreads();
  }

  // ---- epilogue: scale, bias, activation, optional strict-upper mask ----
#pragma unroll
  for (int mi = 0; mi < 2; ++mi) {
#pragma unroll
    for (int ni = 0; ni < 4; ++ni) {
      int colg = colBase + wcol * 64 + ni * 16 + lr;
      if (colg >= Ncols) continue;
      float bv = bias ? bias[colg] : 0.f;
#pragma unroll
      for (int v = 0; v < 8; ++v) {
        int rowg = rowBase + wrow * 32 + mi * 16 + v + 8 * lh;
        if (rowg >= Mrows) continue;
        float x = c[mi][ni][v] * scale + bv;
        if (act == 1)      x = gelu_exact(x);
        else if (act == 2) x = fmaxf(x, 0.f);
        else if (act == 3) x = 1.f / (1.f + __expf(-x));
        if (strictUpper && colg <= rowg) x = 0.f;     // triu(k=1) mask
        Ob[(long long)rowg * ldo + colg] = x;
      }
    }
  }
}

// ---------------------------------------------------------------------------
// Structure-exploiting kernels
// ---------------------------------------------------------------------------

// e_pre[b,i,:] = (sum_{t in window i} w(t) * f_fg[b,t,:]) / colsum(i)
__global__ void pool_events_kernel(const float* __restrict__ f_fg,
                                   float* __restrict__ e_pre) {
  int i = blockIdx.x, b = blockIdx.y;
  float cs = 0.f;
  for (int dt = 0; dt < 8; ++dt) {
    int t = 4 * i + dt;
    if (t < T_) cs += 1.f / (float)(row_hi(t) - row_lo(t) + 1);
  }
  cs = fmaxf(cs, 1.f);
  float inv = 1.f / cs;
  for (int d = threadIdx.x; d < DIM; d += 256) {
    float acc = 0.f;
    for (int dt = 0; dt < 8; ++dt) {
      int t = 4 * i + dt;
      if (t < T_) {
        float w = 1.f / (float)(row_hi(t) - row_lo(t) + 1);
        acc += w * f_fg[((long long)b * T_ + t) * DIM + d];
      }
    }
    e_pre[((long long)b * M_ + i) * DIM + d] = acc * inv;
  }
}

// e += sinusoidal positional encoding over events
__global__ void add_pe_kernel(float* __restrict__ e) {
  int i = blockIdx.x, b = blockIdx.y;
  for (int d = threadIdx.x; d < DIM; d += 256) {
    float two_j = (float)(d & ~1);
    float div = __expf(-two_j * 9.210340371976184f / (float)DIM); // ln(10000)
    float ang = (float)i * div;
    float pev = (d & 1) ? cosf(ang) : sinf(ang);
    e[((long long)b * M_ + i) * DIM + d] += pev;
  }
}

// h_frame_pre[b,t,:] = sum_{|s-t|<8} (a_temp[t,s]/rowsum) * f_fg[b,s,:]
__global__ void frame_band_kernel(const float* __restrict__ f_fg,
                                  float* __restrict__ outp /* ld = 2*DIM */) {
  int t = blockIdx.x, b = blockIdx.y;
  float w[15]; float rs = 0.f;
  for (int dsi = -7; dsi <= 7; ++dsi) {
    int s = t + dsi;
    float wv = 0.f;
    if (s >= 0 && s < T_)
      wv = (dsi == 0) ? 1.f : __expf(-(float)(dsi * dsi) / 18.f); // 2*sigma^2=18
    w[dsi + 7] = wv; rs += wv;
  }
  float inv = 1.f / rs;
  for (int d = threadIdx.x; d < DIM; d += 256) {
    float acc = 0.f;
#pragma unroll
    for (int j = 0; j < 15; ++j) {
      if (w[j] != 0.f)
        acc += w[j] * f_fg[((long long)b * T_ + (t - 7 + j)) * DIM + d];
    }
    outp[((long long)b * T_ + t) * (2 * DIM) + d] = acc * inv;
  }
}

// a_temp_b[b,t,:] analytic broadcast write
__global__ void write_atemp_kernel(float* __restrict__ out) {
  int t = blockIdx.x, b = blockIdx.y;
  float* row = out + ((long long)b * T_ + t) * T_;
  for (int s = threadIdx.x; s < T_; s += 256) {
    int ds = s - t, ad = ds < 0 ? -ds : ds;
    float v = 0.f;
    if (ad == 0) v = 1.f;
    else if (ad < 8) v = __expf(-(float)(ds * ds) / 18.f);
    row[s] = v;
  }
}

// mem_b[b,t,:] analytic broadcast write
__global__ void write_memb_kernel(float* __restrict__ out) {
  int t = blockIdx.x, b = blockIdx.y;
  int lo = row_lo(t), hi = row_hi(t);
  float wv = 1.f / (float)(hi - lo + 1);
  float* row = out + ((long long)b * T_ + t) * M_;
  for (int i = threadIdx.x; i < M_; i += 256)
    row[i] = (i >= lo && i <= hi) ? wv : 0.f;
}

// escale[b,m] = 1 / max(||e[b,m]||, 1e-12)
__global__ void row_l2_invnorm_kernel(const float* __restrict__ e,
                                      float* __restrict__ outS) {
  int m = blockIdx.x, b = blockIdx.y;
  const float* p = e + ((long long)b * M_ + m) * DIM;
  __shared__ float red[256];
  float ss = 0.f;
  for (int d = threadIdx.x; d < DIM; d += 256) { float v = p[d]; ss += v * v; }
  red[threadIdx.x] = ss; __syncthreads();
  for (int s = 128; s > 0; s >>= 1) {
    if (threadIdx.x < s) red[threadIdx.x] += red[threadIdx.x + s];
    __syncthreads();
  }
  if (threadIdx.x == 0)
    outS[(long long)b * M_ + m] = 1.f / fmaxf(sqrtf(red[0]), 1e-12f);
}

// mix MLP stage 1: h[b,j] = gelu(f_q[b,:] @ w1[:,j] + b1[j])
__global__ void mix_hidden_kernel(const float* __restrict__ fq,
                                  const float* __restrict__ w1,
                                  const float* __restrict__ b1,
                                  float* __restrict__ h) {
  int b = blockIdx.y;
  int j = blockIdx.x * 256 + threadIdx.x;
  __shared__ float q[DIM];
  for (int d = threadIdx.x; d < DIM; d += 256) q[d] = fq[(long long)b * DIM + d];
  __syncthreads();
  if (j < DIM) {
    float acc = b1[j];
    for (int d = 0; d < DIM; ++d) acc += q[d] * w1[(long long)d * DIM + j];
    h[(long long)b * DIM + j] = gelu_exact(acc);
  }
}

// mix MLP stage 2 + softmax -> lam[b,2]
__global__ void mix_logits_kernel(const float* __restrict__ h,
                                  const float* __restrict__ w2,
                                  const float* __restrict__ b2,
                                  float* __restrict__ lam) {
  int b = blockIdx.x;
  __shared__ float r0[256], r1[256];
  float a0 = 0.f, a1 = 0.f;
  for (int j = threadIdx.x; j < DIM; j += 256) {
    float hv = h[(long long)b * DIM + j];
    a0 += hv * w2[2 * j]; a1 += hv * w2[2 * j + 1];
  }
  r0[threadIdx.x] = a0; r1[threadIdx.x] = a1; __syncthreads();
  for (int s = 128; s > 0; s >>= 1) {
    if (threadIdx.x < s) { r0[threadIdx.x] += r0[threadIdx.x + s]; r1[threadIdx.x] += r1[threadIdx.x + s]; }
    __syncthreads();
  }
  if (threadIdx.x == 0) {
    float l0 = r0[0] + b2[0], l1 = r1[0] + b2[1];
    float mx = fmaxf(l0, l1);
    float e0 = __expf(l0 - mx), e1 = __expf(l1 - mx);
    float inv = 1.f / (e0 + e1);
    lam[2 * b] = e0 * inv; lam[2 * b + 1] = e1 * inv;
  }
}

// a_event = lam0 * a_sem + lam1 * a_cau
__global__ void combine_aevent_kernel(const float* __restrict__ asem,
                                      const float* __restrict__ acau,
                                      const float* __restrict__ lam,
                                      float* __restrict__ out) {
  int b = blockIdx.y;
  long long idx = (long long)blockIdx.x * 256 + threadIdx.x;
  if (idx < (long long)M_ * M_) {
    long long o = (long long)b * M_ * M_ + idx;
    out[o] = lam[2 * b] * asem[o] + lam[2 * b + 1] * acau[o];
  }
}

// evrow[b,m] = 1 / clip(sum_n a_event[b,m,n], 1e-6)
__global__ void aev_row_invsum_kernel(const float* __restrict__ a,
                                      float* __restrict__ outS) {
  int m = blockIdx.x, b = blockIdx.y;
  const float* p = a + ((long long)b * M_ + m) * M_;
  __shared__ float red[256];
  float s = 0.f;
  for (int n = threadIdx.x; n < M_; n += 256) s += p[n];
  red[threadIdx.x] = s; __syncthreads();
  for (int st = 128; st > 0; st >>= 1) {
    if (threadIdx.x < st) red[threadIdx.x] += red[threadIdx.x + st];
    __syncthreads();
  }
  if (threadIdx.x == 0)
    outS[(long long)b * M_ + m] = 1.f / fmaxf(red[0], 1e-6f);
}

// indeg[b,n] = sum_m a_event[b,m,n]
__global__ void aev_colsum_kernel(const float* __restrict__ a,
                                  float* __restrict__ indeg) {
  int col = blockIdx.x * 256 + threadIdx.x;
  int b = blockIdx.y;
  if (col < M_) {
    const float* p = a + (long long)b * M_ * M_ + col;
    float s = 0.f;
    for (int m = 0; m < M_; ++m) s += p[(long long)m * M_];
    indeg[(long long)b * M_ + col] = s;
  }
}

// concat[b,t,DIM + d] = sum_{i in members(t)} w(t) * h_event[b,i,d]
__global__ void he2f_kernel(const float* __restrict__ h_event,
                            float* __restrict__ concatb) {
  int t = blockIdx.x, b = blockIdx.y;
  int lo = row_lo(t), hi = row_hi(t);
  float wv = 1.f / (float)(hi - lo + 1);
  for (int d = threadIdx.x; d < DIM; d += 256) {
    float acc = 0.f;
    for (int i = lo; i <= hi; ++i)
      acc += h_event[((long long)b * M_ + i) * DIM + d];
    concatb[((long long)b * T_ + t) * (2 * DIM) + DIM + d] = wv * acc;
  }
}

// s_graph[b,t] = 0.5*s_sem + 0.5*sum_i mem[t,i]*indeg[b,i]
__global__ void sgraph_kernel(const float* __restrict__ s_sem,
                              const float* __restrict__ indeg,
                              float* __restrict__ outp) {
  long long idx = (long long)blockIdx.x * 256 + threadIdx.x;
  if (idx >= (long long)B_ * T_) return;
  int b = (int)(idx / T_), t = (int)(idx % T_);
  int lo = row_lo(t), hi = row_hi(t);
  float wv = 1.f / (float)(hi - lo + 1);
  float acc = 0.f;
  for (int i = lo; i <= hi; ++i) acc += indeg[(long long)b * M_ + i];
  outp[idx] = 0.5f * s_sem[idx] + 0.5f * wv * acc;
}

// ---------------------------------------------------------------------------
// Host orchestration
// ---------------------------------------------------------------------------
static inline void launch_gemm(hipStream_t st,
    const float* A, long long sA, int lda,
    const float* Bm, long long sB, int ldb, int transB,
    const float* bias, const float* rowscale, long long sRS,
    const float* colscale, long long sCS,
    float* out, long long sOut, int ldo,
    int Mrows, int K, int Ncols, int nb,
    float scale, int act, int strictUpper)
{
  dim3 grid((Ncols + BLK_N - 1) / BLK_N, (Mrows + BLK_M - 1) / BLK_M, nb);
  wmma_gemm_kernel<<<grid, 256, 0, st>>>(A, sA, lda, Bm, sB, ldb, transB,
      bias, rowscale, sRS, colscale, sCS, out, sOut, ldo,
      Mrows, K, Ncols, scale, act, strictUpper);
}

extern "C" void kernel_launch(void* const* d_in, const int* in_sizes, int n_in,
                              void* d_out, int out_size, void* d_ws, size_t ws_size,
                              hipStream_t stream) {
  (void)in_sizes; (void)n_in; (void)out_size; (void)ws_size;
  // inputs: f_fg, s_sem, f_q, then params dict leaves in insertion order
  const float* f_fg    = (const float*)d_in[0];
  const float* s_sem   = (const float*)d_in[1];
  const float* f_q     = (const float*)d_in[2];
  const float* pool_w1 = (const float*)d_in[3];
  const float* pool_b1 = (const float*)d_in[4];
  const float* pool_w2 = (const float*)d_in[5];
  const float* pool_b2 = (const float*)d_in[6];
  const float* wq      = (const float*)d_in[7];
  const float* wk      = (const float*)d_in[8];
  const float* mix_w1  = (const float*)d_in[9];
  const float* mix_b1  = (const float*)d_in[10];
  const float* mix_w2  = (const float*)d_in[11];
  const float* mix_b2  = (const float*)d_in[12];
  const float* fr_w1   = (const float*)d_in[13];
  const float* fr_b1   = (const float*)d_in[14];
  const float* fr_w2   = (const float*)d_in[15];
  const float* fr_b2   = (const float*)d_in[16];
  const float* ev_w1   = (const float*)d_in[17];
  const float* ev_b1   = (const float*)d_in[18];
  const float* ev_w2   = (const float*)d_in[19];
  const float* ev_b2   = (const float*)d_in[20];
  const float* fu_w1   = (const float*)d_in[21];
  const float* fu_b1   = (const float*)d_in[22];
  const float* fu_w2   = (const float*)d_in[23];
  const float* fu_b2   = (const float*)d_in[24];

  float* out = (float*)d_out;
  // output regions (return order): h_graph, e, a_temp_b, a_event, s_graph, mem_b
  long long o_e  = (long long)B_ * T_ * DIM;
  long long o_at = o_e  + (long long)B_ * M_ * DIM;
  long long o_ae = o_at + (long long)B_ * T_ * T_;
  long long o_sg = o_ae + (long long)B_ * M_ * M_;
  long long o_mb = o_sg + (long long)B_ * T_;
  float* h_graph = out;
  float* e_out   = out + o_e;
  float* atemp   = out + o_at;
  float* aevent  = out + o_ae;
  float* sgraph  = out + o_sg;
  float* memb    = out + o_mb;

  // workspace layout (floats)
  float* ws     = (float*)d_ws;
  float* tmpH   = ws;                                        // B*T*DIM (MLP hidden)
  float* concatb= tmpH    + (long long)B_ * T_ * DIM;        // B*T*2*DIM
  float* epre   = concatb + (long long)B_ * T_ * 2 * DIM;    // B*M*DIM (also h_event_pre)
  float* qbuf   = epre    + (long long)B_ * M_ * DIM;        // also h_event
  float* kbuf   = qbuf    + (long long)B_ * M_ * DIM;
  float* asem   = kbuf    + (long long)B_ * M_ * DIM;        // B*M*M
  float* acau   = asem    + (long long)B_ * M_ * M_;         // B*M*M
  float* escale = acau    + (long long)B_ * M_ * M_;         // B*M
  float* evrow  = escale  + (long long)B_ * M_;              // B*M
  float* indeg  = evrow   + (long long)B_ * M_;              // B*M
  float* mixh   = indeg   + (long long)B_ * M_;              // B*DIM
  float* lam    = mixh    + (long long)B_ * DIM;             // B*2

  const int BM = B_ * M_;         // 4088 flattened event-rows
  const int BT = B_ * T_;         // 16384 flattened frame-rows
  const float inv_sqrt_d = 1.0f / sqrtf((float)DIM);

  // 1) event pooling -> e_pre
  pool_events_kernel<<<dim3(M_, B_), 256, 0, stream>>>(f_fg, epre);

  // 2) pool MLP (shared weights -> flatten batch): e = mlp(e_pre) ; then + PE
  launch_gemm(stream, epre, 0, DIM, pool_w1, 0, DIM, 0, pool_b1,
              nullptr, 0, nullptr, 0, tmpH, 0, DIM, BM, DIM, DIM, 1, 1.f, 1, 0);
  launch_gemm(stream, tmpH, 0, DIM, pool_w2, 0, DIM, 0, pool_b2,
              nullptr, 0, nullptr, 0, e_out, 0, DIM, BM, DIM, DIM, 1, 1.f, 0, 0);
  add_pe_kernel<<<dim3(M_, B_), 256, 0, stream>>>(e_out);

  // 3) q = e@wq, k = e@wk
  launch_gemm(stream, e_out, 0, DIM, wq, 0, DIM, 0, nullptr,
              nullptr, 0, nullptr, 0, qbuf, 0, DIM, BM, DIM, DIM, 1, 1.f, 0, 0);
  launch_gemm(stream, e_out, 0, DIM, wk, 0, DIM, 0, nullptr,
              nullptr, 0, nullptr, 0, kbuf, 0, DIM, BM, DIM, DIM, 1, 1.f, 0, 0);

  // 4) a_sem = relu(e_norm @ e_norm^T)   (per-batch Gram via NT + row/col scales)
  row_l2_invnorm_kernel<<<dim3(M_, B_), 256, 0, stream>>>(e_out, escale);
  launch_gemm(stream, e_out, (long long)M_ * DIM, DIM,
              e_out, (long long)M_ * DIM, DIM, 1, nullptr,
              escale, M_, escale, M_,
              asem, (long long)M_ * M_, M_, M_, DIM, M_, B_, 1.f, 2, 0);

  // 5) a_cau = sigmoid(q k^T / sqrt(d)) * triu(k=1)
  launch_gemm(stream, qbuf, (long long)M_ * DIM, DIM,
              kbuf, (long long)M_ * DIM, DIM, 1, nullptr,
              nullptr, 0, nullptr, 0,
              acau, (long long)M_ * M_, M_, M_, DIM, M_, B_, inv_sqrt_d, 3, 1);

  // 6) lam = softmax(mix_mlp(f_q))
  mix_hidden_kernel<<<dim3(3, B_), 256, 0, stream>>>(f_q, mix_w1, mix_b1, mixh);
  mix_logits_kernel<<<B_, 256, 0, stream>>>(mixh, mix_w2, mix_b2, lam);

  // 7) a_event = lam0*a_sem + lam1*a_cau ; row-sum / in-degree reductions
  combine_aevent_kernel<<<dim3((M_ * M_ + 255) / 256, B_), 256, 0, stream>>>(
      asem, acau, lam, aevent);
  aev_row_invsum_kernel<<<dim3(M_, B_), 256, 0, stream>>>(aevent, evrow);
  aev_colsum_kernel<<<dim3((M_ + 255) / 256, B_), 256, 0, stream>>>(aevent, indeg);

  // 8) h_event = mlp(norm_adj(a_event) @ e)  (per-batch NN GEMM, row scaled)
  launch_gemm(stream, aevent, (long long)M_ * M_, M_,
              e_out, (long long)M_ * DIM, DIM, 0, nullptr,
              evrow, M_, nullptr, 0,
              epre, (long long)M_ * DIM, DIM, M_, M_, DIM, B_, 1.f, 0, 0);
  launch_gemm(stream, epre, 0, DIM, ev_w1, 0, DIM, 0, ev_b1,
              nullptr, 0, nullptr, 0, tmpH, 0, DIM, BM, DIM, DIM, 1, 1.f, 1, 0);
  launch_gemm(stream, tmpH, 0, DIM, ev_w2, 0, DIM, 0, ev_b2,
              nullptr, 0, nullptr, 0, qbuf, 0, DIM, BM, DIM, DIM, 1, 1.f, 0, 0);

  // 9) h_frame = mlp(norm_adj(a_temp) @ f_fg)  (banded aggregation)
  //    stage h_frame_pre in concat[:,DIM:2*DIM] (overwritten later by h_e2f)
  frame_band_kernel<<<dim3(T_, B_), 256, 0, stream>>>(f_fg, concatb + DIM);
  launch_gemm(stream, concatb + DIM, 0, 2 * DIM, fr_w1, 0, DIM, 0, fr_b1,
              nullptr, 0, nullptr, 0, tmpH, 0, DIM, BT, DIM, DIM, 1, 1.f, 1, 0);
  launch_gemm(stream, tmpH, 0, DIM, fr_w2, 0, DIM, 0, fr_b2,
              nullptr, 0, nullptr, 0, concatb, 0, 2 * DIM, BT, DIM, DIM, 1, 1.f, 0, 0);

  // 10) h_e2f into concat[:,DIM:2*DIM], then fusion MLP -> h_graph
  he2f_kernel<<<dim3(T_, B_), 256, 0, stream>>>(qbuf, concatb);
  launch_gemm(stream, concatb, 0, 2 * DIM, fu_w1, 0, DIM, 0, fu_b1,
              nullptr, 0, nullptr, 0, tmpH, 0, DIM, BT, 2 * DIM, DIM, 1, 1.f, 1, 0);
  launch_gemm(stream, tmpH, 0, DIM, fu_w2, 0, DIM, 0, fu_b2,
              nullptr, 0, nullptr, 0, h_graph, 0, DIM, BT, DIM, DIM, 1, 1.f, 0, 0);

  // 11) analytic broadcast outputs + s_graph
  write_atemp_kernel<<<dim3(T_, B_), 256, 0, stream>>>(atemp);
  write_memb_kernel<<<dim3(T_, B_), 256, 0, stream>>>(memb);
  sgraph_kernel<<<(BT + 255) / 256, 256, 0, stream>>>(s_sem, indeg, sgraph);
}
